// CDEPdeLayerM2_18880676233737
// MI455X (gfx1250) — compile-verified, matrix operand order checked by reading, not run on hardware
//
#include <hip/hip_runtime.h>

// Problem constants (from reference setup_inputs)
#define Bb    4
#define CIN   8
#define COUT  16
#define ORN   8
#define Hh    128
#define Ww    128
#define HW    (Hh * Ww)          // 16384
#define OHW   (ORN * HW)         // 131072
#define NROWS (Bb * OHW)         // 524288 rows of the einsum
#define UELEMS (Bb * CIN * OHW)  // 4194304
#define YELEMS (Bb * COUT * OHW) // 8388608
#define TWO_PI_F 6.28318530717958647692f
#define ALPHA_F 0.65f

typedef __attribute__((ext_vector_type(2)))  float    v2f;
typedef __attribute__((ext_vector_type(8)))  float    v8f;
typedef __attribute__((ext_vector_type(16))) _Float16 v16h;

// Border-clamped bilinear sample of a 128x128 slice.
__device__ __forceinline__ float bilerp(const float* __restrict__ s, float y, float x) {
    float yf = floorf(y), xf = floorf(x);
    float fy = y - yf,    fx = x - xf;
    int y0 = (int)yf,     x0 = (int)xf;
    int y0c = min(max(y0,     0), Hh - 1);
    int y1c = min(max(y0 + 1, 0), Hh - 1);
    int x0c = min(max(x0,     0), Ww - 1);
    int x1c = min(max(x0 + 1, 0), Ww - 1);
    const float* r0 = s + (y0c << 7);
    const float* r1 = s + (y1c << 7);
    float v00 = r0[x0c], v01 = r0[x1c];
    float v10 = r1[x0c], v11 = r1[x1c];
    float a = v00 + fx * (v01 - v00);
    float b = v10 + fx * (v11 - v10);
    return a + fy * (b - a);
}

// ---------------------------------------------------------------------------
// Kernel 1: convection  u1[b,c,o,h,w] = resample of x under per-channel g0
// ---------------------------------------------------------------------------
__global__ void conv_kernel(const float* __restrict__ x,
                            const float* __restrict__ g0,
                            float* __restrict__ out) {
    int idx = blockIdx.x * blockDim.x + threadIdx.x;
    if (idx >= UELEMS) return;
    int hw = idx & (HW - 1);
    int t  = idx >> 14;
    int o  = t & 7;  t >>= 3;
    int c  = t & 7;
    int b  = t >> 3;
    float h = (float)(hw >> 7), w = (float)(hw & 127);

    float x0  = g0[c * 3 + 0];
    float y0  = g0[c * 3 + 1];
    float th0 = g0[c * 3 + 2];

    // fractional orientation roll: v[o] = (1-f)*u[(o-i)&7] + f*u[(o-i-1)&7]
    float tt = th0 * ((float)ORN / TWO_PI_F);
    float fi = floorf(tt);
    float f  = tt - fi;
    int   i  = (int)fi;
    int o0 = (o - i) & 7;
    int o1 = (o0 + 7) & 7;

    float sa, ca;
    sincosf((float)o * (TWO_PI_F / ORN) - th0, &sa, &ca);
    float dx = -(ca * x0 - sa * y0);
    float dy = -(sa * x0 + ca * y0);

    const float* base = x + (size_t)(b * CIN + c) * OHW;
    float s0 = bilerp(base + o0 * HW, h + dy, w + dx);
    float s1 = bilerp(base + o1 * HW, h + dy, w + dx);
    out[idx] = (1.0f - f) * s0 + f * s1;
}

// ---------------------------------------------------------------------------
// Kernel 2: fractional morphological dilation (DIL=1) / erosion (DIL=0)
// 27 taps over (qt,qy,qx); k(c,tap) and rotated offsets cached in LDS.
// ---------------------------------------------------------------------------
template <int DIL>
__global__ void morph_kernel(const float* __restrict__ u,
                             const float* __restrict__ metric,
                             float* __restrict__ out) {
    __shared__ float kk[CIN * 27];     // kernel values per (c, tap27)
    __shared__ float dxs[ORN * 9];     // rotated offsets per (o, tap9)
    __shared__ float dys[ORN * 9];

    int tid = threadIdx.x;
    if (tid < CIN * 27) {
        int c   = tid / 27, tap = tid % 27;
        int qt  = tap / 9 - 1;
        int qy  = (tap % 9) / 3 - 1;
        int qx  = tap % 3 - 1;
        float dth = (float)qt * (TWO_PI_F / ORN);
        float c1, c2, c3;
        if (qt == 0) { c1 = (float)qx; c2 = (float)qy; c3 = 0.0f; }
        else {
            float hh  = 0.5f * dth;
            float cot = cosf(hh) / sinf(hh);
            c1 = hh * ((float)qx * cot + (float)qy);
            c2 = hh * (-(float)qx + (float)qy * cot);
            c3 = dth;
        }
        float m0 = metric[c * 3 + 0], m1 = metric[c * 3 + 1], m2 = metric[c * 3 + 2];
        float s  = (m0 * c1) * (m0 * c1) + (m1 * c2) * (m1 * c2) + (m2 * c3) * (m2 * c3);
        const float P  = 2.0f * ALPHA_F / (2.0f * ALPHA_F - 1.0f);      // 4.333...
        const float NU = (2.0f * ALPHA_F - 1.0f) * powf(2.0f * ALPHA_F, -P);
        kk[tid] = NU * powf(s, P * 0.5f);
    }
    if (tid < ORN * 9) {
        int o   = tid / 9, t9 = tid % 9;
        int qy  = t9 / 3 - 1, qx = t9 % 3 - 1;
        float sa, ca;
        sincosf((float)o * (TWO_PI_F / ORN), &sa, &ca);
        dxs[tid] = ca * (float)qx - sa * (float)qy;
        dys[tid] = sa * (float)qx + ca * (float)qy;
    }
    __syncthreads();

    int idx = blockIdx.x * blockDim.x + tid;
    if (idx >= UELEMS) return;
    int hw = idx & (HW - 1);
    int t  = idx >> 14;
    int o  = t & 7;  t >>= 3;
    int c  = t & 7;
    int b  = t >> 3;
    float h = (float)(hw >> 7), w = (float)(hw & 127);

    const float* ubc = u + (size_t)(b * CIN + c) * OHW;
    float acc = DIL ? -INFINITY : INFINITY;
    #pragma unroll
    for (int qt = -1; qt <= 1; ++qt) {
        const float* slice = ubc + ((o + qt) & 7) * HW;
        #pragma unroll
        for (int t9 = 0; t9 < 9; ++t9) {
            float val = bilerp(slice, h + dys[o * 9 + t9], w + dxs[o * 9 + t9]);
            float k   = kk[c * 27 + (qt + 1) * 9 + t9];
            float cand = DIL ? (val - k) : (val + k);
            acc = DIL ? fmaxf(acc, cand) : fminf(acc, cand);
        }
    }
    out[idx] = acc;
}

// ---------------------------------------------------------------------------
// Kernel 3: einsum bcohw,cd->bdohw via WMMA + BN statistics (sum / sumsq)
// One wave per 16-row x 16-col tile. K=8: two chained f32 16x16x4 WMMAs
// (exact f32), or one zero-padded f16 16x16x32 WMMA as fallback.
// ---------------------------------------------------------------------------
__global__ void gemm_bnstat_kernel(const float* __restrict__ u,
                                   const float* __restrict__ W,
                                   float* __restrict__ y,
                                   float* __restrict__ stats) {
    __shared__ float ssum[COUT];
    __shared__ float ssq[COUT];
    int tid = threadIdx.x;
    if (tid < COUT) { ssum[tid] = 0.0f; ssq[tid] = 0.0f; }
    __syncthreads();

    int wave = tid >> 5;
    int lane = tid & 31;
    int tile = blockIdx.x * 8 + wave;   // 16 rows per tile
    int m    = lane & 15;

    v8f acc = {};
#if __has_builtin(__builtin_amdgcn_wmma_f32_16x16x4_f32)
    // f32 A 16x4 layout: lanes 0-15 hold K=0,1; lanes 16-31 hold K=2,3 (M = lane&15)
    {
        int   kof = (lane < 16) ? 0 : 2;
        int   row = tile * 16 + m;           // row index in [0, NROWS)
        int   bb  = row >> 17;               // / OHW
        int   rr  = row & (OHW - 1);
        const float* up = u + (size_t)bb * CIN * OHW + rr;
        v2f a0, a1, b0, b1;
        a0[0] = up[(size_t)(kof + 0) * OHW]; a0[1] = up[(size_t)(kof + 1) * OHW];
        a1[0] = up[(size_t)(kof + 4) * OHW]; a1[1] = up[(size_t)(kof + 5) * OHW];
        b0[0] = W[(kof + 0) * COUT + m];     b0[1] = W[(kof + 1) * COUT + m];
        b1[0] = W[(kof + 4) * COUT + m];     b1[1] = W[(kof + 5) * COUT + m];
        acc = __builtin_amdgcn_wmma_f32_16x16x4_f32(false, a0, false, b0,
                                                    (short)0, acc, false, false);
        acc = __builtin_amdgcn_wmma_f32_16x16x4_f32(false, a1, false, b1,
                                                    (short)0, acc, false, false);
    }
#else
    // f16 fallback: K=8 real data zero-padded to K=32
    {
        v16h av = {};
        v16h bv = {};
        if (lane < 16) {
            int   row = tile * 16 + lane;
            int   bb  = row >> 17;
            int   rr  = row & (OHW - 1);
            const float* up = u + (size_t)bb * CIN * OHW + rr;
            #pragma unroll
            for (int c = 0; c < CIN; ++c) {
                av[c] = (_Float16)up[(size_t)c * OHW];
                bv[c] = (_Float16)W[c * COUT + lane];
            }
        }
        acc = __builtin_amdgcn_wmma_f32_16x16x32_f16(false, av, false, bv,
                                                     (short)0, acc, false, false);
    }
#endif

    // D layout: lane<16 -> N=lane, VGPR r = row M=r; lane>=16 -> M=r+8.
    int d     = m;
    int mBase = (lane < 16) ? 0 : 8;
    int bb2   = (tile * 16) >> 17;
    int rr2   = (tile * 16) & (OHW - 1);
    float* yp = y + (size_t)bb2 * COUT * OHW + (size_t)d * OHW + rr2 + mBase;
    float s = 0.0f, s2 = 0.0f;
    #pragma unroll
    for (int mm = 0; mm < 8; ++mm) {
        float v = acc[mm];
        yp[mm] = v;
        s  += v;
        s2 += v * v;
    }
    atomicAdd(&ssum[d], s);
    atomicAdd(&ssq[d],  s2);
    __syncthreads();
    if (tid < COUT) {
        atomicAdd(&stats[tid],        ssum[tid]);
        atomicAdd(&stats[COUT + tid], ssq[tid]);
    }
}

// ---------------------------------------------------------------------------
// Kernel 4: BatchNorm finalize (in-place on y)
// ---------------------------------------------------------------------------
__global__ void bn_finalize_kernel(float* __restrict__ y,
                                   const float* __restrict__ stats,
                                   const float* __restrict__ gamma,
                                   const float* __restrict__ beta) {
    int idx = blockIdx.x * blockDim.x + threadIdx.x;
    if (idx >= YELEMS) return;
    int d = (idx >> 17) & (COUT - 1);   // (idx / OHW) % COUT
    float invN = 1.0f / (float)NROWS;
    float mean = stats[d] * invN;
    float var  = stats[COUT + d] * invN - mean * mean;
    float inv  = rsqrtf(var + 1e-5f);
    y[idx] = (y[idx] - mean) * inv * gamma[d] + beta[d];
}

// ---------------------------------------------------------------------------
extern "C" void kernel_launch(void* const* d_in, const int* in_sizes, int n_in,
                              void* d_out, int out_size, void* d_ws, size_t ws_size,
                              hipStream_t stream) {
    (void)in_sizes; (void)n_in; (void)out_size; (void)ws_size;
    const float* x          = (const float*)d_in[0];
    const float* g0         = (const float*)d_in[1];
    const float* metric_dil = (const float*)d_in[2];
    const float* metric_ero = (const float*)d_in[3];
    const float* weight     = (const float*)d_in[4];
    const float* gamma      = (const float*)d_in[5];
    const float* beta       = (const float*)d_in[6];
    float* out = (float*)d_out;

    float* buf0  = (float*)d_ws;           // 4194304 floats
    float* buf1  = buf0 + UELEMS;          // 4194304 floats
    float* stats = buf1 + UELEMS;          // 32 floats (sum[16], sumsq[16])

    hipMemsetAsync(stats, 0, 2 * COUT * sizeof(float), stream);

    dim3 blk(256);
    conv_kernel      <<<UELEMS / 256, blk, 0, stream>>>(x, g0, buf0);
    morph_kernel<1>  <<<UELEMS / 256, blk, 0, stream>>>(buf0, metric_dil, buf1);
    morph_kernel<0>  <<<UELEMS / 256, blk, 0, stream>>>(buf1, metric_ero, buf0);
    gemm_bnstat_kernel<<<NROWS / 16 / 8, blk, 0, stream>>>(buf0, weight, out, stats);
    bn_finalize_kernel<<<YELEMS / 256, blk, 0, stream>>>(out, stats, gamma, beta);
}
